// FewShotNCALoss_13606456394198
// MI455X (gfx1250) — matrix operand order repaired and checked by reading, not compiled, not run on hardware
//
#include <hip/hip_runtime.h>

// ---------------------------------------------------------------------------
// FewShotNCALoss on MI455X (gfx1250, wave32)
//   n=4096, d=1024. Dominant cost: pred @ pred^T (34.4 GFLOP GEMM).
//   All working data fits in the 192MB L2 -> compute bound -> bf16 WMMA
//   (v_wmma_f32_16x16x32_bf16), Gram tiles consumed in registers with a fused
//   exp/mask/column-sum epilogue; the 64MB Gram matrix is never materialized.
//   Staging uses the CDNA5 async copy engine: GLOBAL_LOAD_ASYNC_TO_LDS_B128
//   with triple-buffered LDS, ASYNCcnt-gated, one barrier per K-step.
//   B fragments are batch-loaded (16x ds_load_b128) before the 8-WMMA burst
//   so the matrix pipe issues back-to-back with partial dscnt waits.
// Workspace (~8.05 MB):
//   [0, 8MB)   : pred in bf16 (4096x1024)
//   +8MB       : sq[4096]  row squared norms
//   +16KB      : pos[4096] column sums of exp(+p/T) over same-class i!=j
//   +16KB      : neg[4096] column sums of exp(-p/T) over diff-class
// ---------------------------------------------------------------------------

typedef unsigned short u16;
typedef __attribute__((ext_vector_type(16))) __bf16 v16bf;
typedef __attribute__((ext_vector_type(8)))  float  v8f;

#define N_SAMP 4096
#define N_DIM  1024
#define NK     (N_DIM / 32)
#define EPSF   1e-10f

static __device__ __forceinline__ u16 f2bf(float f) {
    unsigned int x = __float_as_uint(f);
    x += 0x7fffu + ((x >> 16) & 1u);      // round-to-nearest-even
    return (u16)(x >> 16);
}

// --------------------------- kernel 1: prep --------------------------------
__global__ __launch_bounds__(256)
void nca_prep(const float* __restrict__ pred, u16* __restrict__ predB,
              float* __restrict__ sq, float* __restrict__ pos,
              float* __restrict__ neg) {
    const int row = blockIdx.x, tid = threadIdx.x;
    const float4 v = *(const float4*)(pred + (size_t)row * N_DIM + tid * 4);
    unsigned int lo = (unsigned int)f2bf(v.x) | ((unsigned int)f2bf(v.y) << 16);
    unsigned int hi = (unsigned int)f2bf(v.z) | ((unsigned int)f2bf(v.w) << 16);
    uint2 w; w.x = lo; w.y = hi;
    *(uint2*)(predB + (size_t)row * N_DIM + tid * 4) = w;

    __shared__ float red[256];
    red[tid] = v.x * v.x + v.y * v.y + v.z * v.z + v.w * v.w;
    __syncthreads();
    for (int o = 128; o > 0; o >>= 1) {
        if (tid < o) red[tid] += red[tid + o];
        __syncthreads();
    }
    if (tid == 0) { sq[row] = red[0]; pos[row] = 0.f; neg[row] = 0.f; }
}

// --------------------------- kernel 2: main --------------------------------
// Grid 32x32; 256 threads (8 waves) own a 128x128 output tile.
// Wave w: rows [rowBase+16w, +16) x 128 cols = 8 accumulators of
// v_wmma_f32_16x16x32_bf16. K in steps of 32; global->LDS via async copy,
// triple-buffered, 2 tiles in flight, one s_barrier per K-step.
__global__ __launch_bounds__(256)
void nca_main(const u16* __restrict__ predB, const float* __restrict__ sq,
              const int* __restrict__ target, const float* __restrict__ tempP,
              float* __restrict__ pos, float* __restrict__ neg) {
    __shared__ __align__(16) u16 lA[3][128 * 32];   // 3 x 8 KB
    __shared__ __align__(16) u16 lB[3][128 * 32];   // 3 x 8 KB
    __shared__ float lpos[128];
    __shared__ float lneg[128];

    const int tid  = threadIdx.x;
    const int wave = tid >> 5, lane = tid & 31;
    const int m    = lane & 15, hi = lane >> 4;
    const int rowBase = blockIdx.y * 128;
    const int colBase = blockIdx.x * 128;

    if (tid < 128) { lpos[tid] = 0.f; lneg[tid] = 0.f; }

    // staging: each thread moves 32B of each 128x32 bf16 tile (2 async b128).
    const int srow    = tid >> 1;        // tile row 0..127
    const int halfsel = tid & 1;         // which 32B half of the 64B row
    const unsigned stageOff = (unsigned)(srow * 64 + halfsel * 32);
    const u16* gA = predB + (size_t)(rowBase + srow) * N_DIM + halfsel * 16;
    const u16* gB = predB + (size_t)(colBase + srow) * N_DIM + halfsel * 16;

    // async issue: IOFFSET is added to BOTH the LDS and global address
    // (ISA 08 4.4), so offset:16 covers the second 16B of our 32B chunk.
    auto issue = [&](int kt, int buf) {
        unsigned la = (unsigned)(unsigned long long)(&lA[buf][0]) + stageOff;
        unsigned lb = (unsigned)(unsigned long long)(&lB[buf][0]) + stageOff;
        unsigned long long ga = (unsigned long long)(gA + kt * 32);
        unsigned long long gb = (unsigned long long)(gB + kt * 32);
        asm volatile("global_load_async_to_lds_b128 %0, %1, off\n\t"
                     "global_load_async_to_lds_b128 %0, %1, off offset:16"
                     :: "v"(la), "v"(ga) : "memory");
        asm volatile("global_load_async_to_lds_b128 %0, %1, off\n\t"
                     "global_load_async_to_lds_b128 %0, %1, off offset:16"
                     :: "v"(lb), "v"(gb) : "memory");
    };

    v8f acc[8] = {};                      // 8 x (16x16 f32) accumulators

    issue(0, 0);
    issue(1, 1);

    for (int kt = 0; kt < NK; ++kt) {
        // Retire own 4 async loads of tile kt (completion is in-order; at most
        // the next tile's 4 may remain). Last tile: drain fully.
        if (kt + 1 < NK) asm volatile("s_wait_asynccnt 4" ::: "memory");
        else             asm volatile("s_wait_asynccnt 0" ::: "memory");
        // One barrier: publishes tile kt to all waves AND proves everyone is
        // done consuming tile kt-1 (so its buffer may be overwritten below).
        __syncthreads();
        if (kt + 2 < NK) issue(kt + 2, (kt + 2) % 3);

        const u16* bufA = lA[kt % 3];
        const u16* bufB = lB[kt % 3];

        // A fragment (16x32 bf16, ISA 7.12.2): lane m: K0..7 | K16..23,
        // lane m+16: K8..15 | K24..31 of row (wave*16 + m).
        union { uint4 u[2]; v16bf v; } af;
        const uint4* qa = (const uint4*)(bufA + (wave * 16 + m) * 32);
        af.u[0] = qa[hi];
        af.u[1] = qa[2 + hi];

        // Batch-load all 8 B fragments (32x16 bf16: lane m = column m, low
        // lanes K0..7|K16..23, high lanes K8..15|K24..31). ds_loads complete in order, so
        // the 8 WMMAs below issue back-to-back behind decreasing dscnt waits.
        union { uint4 u[2]; v16bf v; } bfr[8];
#pragma unroll
        for (int t = 0; t < 8; ++t) {
            const uint4* qb = (const uint4*)(bufB + (t * 16 + m) * 32);
            bfr[t].u[0] = qb[hi];
            bfr[t].u[1] = qb[2 + hi];
        }
#pragma unroll
        for (int t = 0; t < 8; ++t) {
            acc[t] = __builtin_amdgcn_wmma_f32_16x16x32_bf16(
                false, af.v, false, bfr[t].v, (short)0, acc[t], false, false);
        }
    }

    // ---------------- fused epilogue ----------------
    // C/D layout: VGPR r, lanes 0-15 -> M=r, lanes 16-31 -> M=r+8, N=lane%16.
    const float invT   = 1.0f / tempP[0];
    const int   giBase = rowBase + wave * 16 + hi * 8;
    float sqi[8]; int ti[8];
#pragma unroll
    for (int r = 0; r < 8; ++r) {
        sqi[r] = sq[giBase + r];
        ti[r]  = target[giBase + r];
    }

#pragma unroll
    for (int t = 0; t < 8; ++t) {
        const int   gj  = colBase + t * 16 + m;
        const float sqj = sq[gj];
        const int   tj  = target[gj];
        float lp = 0.f, ln = 0.f;
#pragma unroll
        for (int r = 0; r < 8; ++r) {
            float p = fmaxf(sqi[r] + sqj - 2.0f * acc[t][r], EPSF);
            if (ti[r] == tj) {
                if (giBase + r != gj) lp += __expf(p * invT);   // positive pair
            } else {
                ln += __expf(-p * invT);                        // negative pair
            }
        }
        atomicAdd(&lpos[t * 16 + m], lp);   // ds_add_f32
        atomicAdd(&lneg[t * 16 + m], ln);
    }
    __syncthreads();
    if (tid < 128) {
        atomicAdd(&pos[colBase + tid], lpos[tid]);   // global_atomic_add_f32
        atomicAdd(&neg[colBase + tid], lneg[tid]);
    }
}

// --------------------------- kernel 3: loss --------------------------------
__global__ __launch_bounds__(256)
void nca_loss(const float* __restrict__ pos, const float* __restrict__ neg,
              float* __restrict__ out) {
    __shared__ float red[256];
    const int tid = threadIdx.x;
    float s = 0.f;
    for (int j = tid; j < N_SAMP; j += 256) {
        float num  = 1.0f / pos[j];           // exp(-log(sum(dist_m*positives)))
        float den  = fmaxf(neg[j], EPSF);
        float frac = num / (num + den);
        if (frac >= EPSF) s += logf(frac);
    }
    red[tid] = s;
    __syncthreads();
    for (int o = 128; o > 0; o >>= 1) {
        if (tid < o) red[tid] += red[tid + o];
        __syncthreads();
    }
    if (tid == 0) out[0] = -red[0] / (float)N_SAMP;
}

// --------------------------- launcher --------------------------------------
extern "C" void kernel_launch(void* const* d_in, const int* in_sizes, int n_in,
                              void* d_out, int out_size, void* d_ws, size_t ws_size,
                              hipStream_t stream) {
    const float* pred   = (const float*)d_in[0];
    const int*   target = (const int*)d_in[1];   // class ids 0..63
    const float* temp   = (const float*)d_in[2]; // scalar temperature (device)
    float*       out    = (float*)d_out;

    char* ws    = (char*)d_ws;
    u16*  predB = (u16*)ws;                                   // 8 MB bf16
    float* sq   = (float*)(ws + (size_t)N_SAMP * N_DIM * 2);  // +16 KB
    float* pos  = sq + N_SAMP;                                // +16 KB
    float* neg  = pos + N_SAMP;                               // +16 KB

    nca_prep<<<N_SAMP, 256, 0, stream>>>(pred, predB, sq, pos, neg);
    nca_main<<<dim3(N_SAMP / 128, N_SAMP / 128), 256, 0, stream>>>(
        predB, sq, target, temp, pos, neg);
    nca_loss<<<1, 256, 0, stream>>>(pos, neg, out);
}
